// FastAttention_19997367730493
// MI455X (gfx1250) — matrix operand order repaired and checked by reading, not compiled, not run on hardware
//
#include <hip/hip_runtime.h>
#include <hip/hip_bf16.h>
#include <math.h>

// ---------- CDNA5 WMMA types ----------
typedef __attribute__((ext_vector_type(16))) _Float16 v16h;
typedef __attribute__((ext_vector_type(8)))  float    v8f;
typedef __attribute__((ext_vector_type(8)))  _Float16 h8;
typedef __attribute__((ext_vector_type(4)))  _Float16 h4;

#define HEADS    16
#define DIM_HEAD 64
#define DIMM     1024
#define NSEQ     4096
#define BATCH    4
#define NROWS    (BATCH * NSEQ)     // 16384
#define NC_QKV   3072

// ---------- CDNA5 async global->LDS copy (ASYNCcnt path, CDNA5 ISA) ----------
// Each lane copies 16B from its global address directly into LDS (no VGPR data).
__device__ __forceinline__ void async_b128(void* lds_ptr, const void* gptr) {
  const uint32_t loff = (uint32_t)(uintptr_t)lds_ptr;      // LDS byte offset (low 32b of generic)
  const uint64_t ga   = (uint64_t)(uintptr_t)gptr;
  asm volatile("global_load_async_to_lds_b128 %0, %1, off"
               :: "v"(loff), "v"(ga) : "memory");
}
__device__ __forceinline__ void wait_async0() {
  asm volatile("s_wait_asynccnt 0" ::: "memory");
}

// ---------- fragment loaders (CDNA5 wave32 layouts, CDNA5 ISA) ----------
// A (16x32 f16): lanes 0-15 row=lane, halfs = K{0..7,16..23}; lanes 16-31 row=lane-16, K{8..15,24..31}
__device__ __forceinline__ v16h load_fragA(const _Float16* t, int ld, int lane) {
  const int row = lane & 15;
  const int kh  = (lane >> 4) << 3;            // 0 or 8
  const _Float16* p = t + row * ld + kh;
  v16h f;
#pragma unroll
  for (int i = 0; i < 8; ++i) f[i] = p[i];
#pragma unroll
  for (int i = 0; i < 8; ++i) f[8 + i] = p[16 + i];
  return f;
}
// B (32x16 f16, stored N-major in LDS): lanes 0-15 col=lane K{0..7,16..23}; lanes 16-31 col=lane-16 K{8..15,24..31}
__device__ __forceinline__ v16h load_fragB(const _Float16* t, int ld, int lane) {
  const int col = lane & 15;
  const int kh  = (lane >> 4) << 3;            // 0 or 8
  const _Float16* p = t + col * ld + kh;
  v16h f;
#pragma unroll
  for (int i = 0; i < 8; ++i) f[i] = p[i];
#pragma unroll
  for (int i = 0; i < 8; ++i) f[8 + i] = p[16 + i];
  return f;
}

__device__ __forceinline__ v8f wmma16(v16h a, v16h b, v8f c) {
  return __builtin_amdgcn_wmma_f32_16x16x32_f16(false, a, false, b, (short)0, c,
                                                false, false);
}

// ---------- kernel: fp32 -> f16 convert (vectorized 16B loads / 8B stores) ----------
__global__ void cvt_f32_f16(const float4* __restrict__ src, h4* __restrict__ dst,
                            long n4) {
  long i = (long)blockIdx.x * blockDim.x + threadIdx.x;
  const long stride = (long)gridDim.x * blockDim.x;
  for (; i < n4; i += stride) {
    float4 v = src[i];
    h4 o;
    o[0] = (_Float16)v.x; o[1] = (_Float16)v.y;
    o[2] = (_Float16)v.z; o[3] = (_Float16)v.w;
    dst[i] = o;
  }
}

// =====================================================================================
//   Pipelined 128x128x32 f16 WMMA GEMM core (double-buffered LDS, async A staging).
//   8 waves arranged 4(m) x 2(n); each wave owns a 32x64 tile = 2x4 WMMAs per k-step.
// =====================================================================================
#define GEMM_CORE(A_PTR, B_PTR, NCOLS)                                                  \
  constexpr int LDA = 40, LDB = 40; /* 80B rows -> 16B aligned for b128 */              \
  __shared__ _Float16 As[2][128 * LDA];                                                 \
  __shared__ _Float16 Bs[2][128 * LDB];                                                 \
  const int tid  = threadIdx.x;                                                         \
  const int lane = tid & 31;                                                            \
  const int wave = tid >> 5;                                                            \
  const int wm = wave & 3, wn = wave >> 2;                                              \
  const int row0 = blockIdx.x * 128;                                                    \
  const int col0 = blockIdx.y * 128;                                                    \
  v8f acc[2][4] = {};                                                                   \
  h8 br0, br1;                                                                          \
  /* --- staging helpers --- */                                                         \
  auto stageA = [&](int buf, int k0) {                                                  \
    _Pragma("unroll")                                                                   \
    for (int r = 0; r < 2; ++r) {           /* 512 16B chunks / 256 threads */          \
      const int e = tid + r * 256;                                                      \
      const int m = e >> 2, c = e & 3;                                                  \
      async_b128(&As[buf][m * LDA + c * 8],                                             \
                 A_PTR + (size_t)(row0 + m) * DIMM + k0 + c * 8);                       \
    }                                                                                   \
  };                                                                                    \
  auto loadB = [&](int k0) {                 /* 512 16B chunks: b128 into regs */       \
    const int k0a = k0 + (tid >> 4),        c0a = (tid & 15) * 8;                       \
    const int k0b = k0 + ((tid + 256) >> 4) - 16, c0b = (tid & 15) * 8;                 \
    br0 = *(const h8*)(B_PTR + (size_t)k0a * (NCOLS) + col0 + c0a);                     \
    br1 = *(const h8*)(B_PTR + (size_t)(k0b + 16) * (NCOLS) + col0 + c0b);              \
  };                                                                                    \
  auto storeB = [&](int buf) {               /* transpose scatter into N-major LDS */   \
    const int ka = tid >> 4, ca = (tid & 15) * 8;                                       \
    const int kb = ka + 16,  cb = ca;                                                   \
    _Pragma("unroll")                                                                   \
    for (int i = 0; i < 8; ++i) Bs[buf][(ca + i) * LDB + ka] = br0[i];                  \
    _Pragma("unroll")                                                                   \
    for (int i = 0; i < 8; ++i) Bs[buf][(cb + i) * LDB + kb] = br1[i];                  \
  };                                                                                    \
  /* --- prologue: stage tile 0 --- */                                                  \
  stageA(0, 0);                                                                         \
  loadB(0);                                                                             \
  storeB(0);                                                                            \
  int buf = 0;                                                                          \
  for (int k0 = 0; k0 < DIMM; k0 += 32) {                                               \
    wait_async0();                            /* this wave's async A writes done */     \
    __syncthreads();                          /* tile `buf` fully staged for all */     \
    const bool has_next = (k0 + 32) < DIMM;                                             \
    if (has_next) {                                                                     \
      stageA(buf ^ 1, k0 + 32);               /* async: overlaps with WMMA below */     \
      loadB(k0 + 32);                         /* b128 loads in flight during WMMA */    \
    }                                                                                   \
    v16h a0 = load_fragA(&As[buf][(wm * 32) * LDA],      LDA, lane);                    \
    v16h a1 = load_fragA(&As[buf][(wm * 32 + 16) * LDA], LDA, lane);                    \
    _Pragma("unroll")                                                                   \
    for (int t = 0; t < 4; ++t) {                                                       \
      v16h bt = load_fragB(&Bs[buf][(wn * 64 + t * 16) * LDB], LDB, lane);              \
      acc[0][t] = wmma16(a0, bt, acc[0][t]);                                            \
      acc[1][t] = wmma16(a1, bt, acc[1][t]);                                            \
    }                                                                                   \
    if (has_next) storeB(buf ^ 1);            /* waits loads after compute issued */    \
    buf ^= 1;                                                                           \
  }

// ---------- kernel: qkv GEMM  C(16384x3072) = Xh * Wh ; scatter into head-major q/k/v ----------
__global__ __launch_bounds__(256) void qkv_gemm(const _Float16* __restrict__ A,
                                                const _Float16* __restrict__ Bm,
                                                float* __restrict__ Qf,
                                                float* __restrict__ Kf,
                                                float* __restrict__ Vf) {
  GEMM_CORE(A, Bm, NC_QKV)

  // col0 is 128-aligned and q/k/v boundaries are 1024-aligned -> `which` constant per block
  const int which = col0 >> 10;
  float* dst = (which == 0) ? Qf : ((which == 1) ? Kf : Vf);
#pragma unroll
  for (int sm = 0; sm < 2; ++sm)
#pragma unroll
    for (int t = 0; t < 4; ++t) {
      const int ncol  = col0 + wn * 64 + t * 16 + (lane & 15);
      const int head  = (ncol >> 6) & 15;
      const int d     = ncol & 63;
      const int mbase = row0 + wm * 32 + sm * 16 + ((lane >> 4) << 3);
#pragma unroll
      for (int j = 0; j < 8; ++j) {
        const int mr = mbase + j;
        const int b = mr >> 12, ns = mr & (NSEQ - 1);
        dst[(((size_t)(b * HEADS + head) * NSEQ + ns) << 6) + d] = acc[sm][t][j];
      }
    }
}

// ---------- kernel: out = Rh(16384x1024) @ WoutH(1024x1024) + b_out -> fp32 ----------
__global__ __launch_bounds__(256) void out_gemm(const _Float16* __restrict__ A,
                                                const _Float16* __restrict__ Bm,
                                                const float* __restrict__ bout,
                                                float* __restrict__ out) {
  GEMM_CORE(A, Bm, DIMM)

#pragma unroll
  for (int sm = 0; sm < 2; ++sm)
#pragma unroll
    for (int t = 0; t < 4; ++t) {
      const int ncol  = col0 + wn * 64 + t * 16 + (lane & 15);
      const int mbase = row0 + wm * 32 + sm * 16 + ((lane >> 4) << 3);
      const float bias = bout[ncol];
#pragma unroll
      for (int j = 0; j < 8; ++j) {
        const int mr = mbase + j;
        out[(size_t)mr * DIMM + ncol] = acc[sm][t][j] + bias;
      }
    }
}

// ---------- kernel: per-(b,h) global-attention statistics ----------
__global__ __launch_bounds__(256) void attn_stats(const float* __restrict__ Qf,
                                                  const float* __restrict__ Kf,
                                                  const float* __restrict__ wq,
                                                  const float* __restrict__ wk,
                                                  const unsigned char* __restrict__ mask,
                                                  float* __restrict__ gq,
                                                  float* __restrict__ gk) {
  __shared__ float logits[NSEQ];
  __shared__ float red[256];
  __shared__ float wvec[64];
  __shared__ float gvec[64];
  const int tid = threadIdx.x;
  const int bh  = blockIdx.x;
  const int b   = bh >> 4;
  const float NEG = -3.402823466e38f;
  const float* qbase = Qf + (size_t)bh * NSEQ * 64;
  const float* kbase = Kf + (size_t)bh * NSEQ * 64;

  // ===== softmax(q . w_q * scale) and gq = sum p*q =====
  if (tid < 64) wvec[tid] = wq[tid];
  __syncthreads();
  for (int n = tid; n < NSEQ; n += 256) {
    float s = 0.f;
    const float* row = qbase + (size_t)n * 64;
#pragma unroll 16
    for (int d = 0; d < 64; ++d) s += row[d] * wvec[d];
    logits[n] = mask[b * NSEQ + n] ? s * 0.125f : NEG;   // scale = 64^-0.5
  }
  __syncthreads();
  float mx = NEG;
  for (int n = tid; n < NSEQ; n += 256) mx = fmaxf(mx, logits[n]);
  red[tid] = mx; __syncthreads();
  for (int s = 128; s > 0; s >>= 1) { if (tid < s) red[tid] = fmaxf(red[tid], red[tid + s]); __syncthreads(); }
  mx = red[0]; __syncthreads();
  float sum = 0.f;
  for (int n = tid; n < NSEQ; n += 256) { float e = __expf(logits[n] - mx); logits[n] = e; sum += e; }
  red[tid] = sum; __syncthreads();
  for (int s = 128; s > 0; s >>= 1) { if (tid < s) red[tid] += red[tid + s]; __syncthreads(); }
  float inv = 1.f / red[0]; __syncthreads();
  {
    const int d = tid & 63, grp = tid >> 6;
    float a = 0.f;
    for (int n = grp; n < NSEQ; n += 4) a += logits[n] * qbase[(size_t)n * 64 + d];
    red[tid] = a; __syncthreads();
    if (tid < 64) {
      float g = (red[tid] + red[tid + 64] + red[tid + 128] + red[tid + 192]) * inv;
      gvec[tid] = g; gq[bh * 64 + tid] = g;
    }
    __syncthreads();
  }

  // ===== k_logits = k . (w_k (.) gq) * scale ; gk = gq (.) sum p_k * k =====
  if (tid < 64) wvec[tid] = wk[tid] * gvec[tid];
  __syncthreads();
  for (int n = tid; n < NSEQ; n += 256) {
    float s = 0.f;
    const float* row = kbase + (size_t)n * 64;
#pragma unroll 16
    for (int d = 0; d < 64; ++d) s += row[d] * wvec[d];
    logits[n] = mask[b * NSEQ + n] ? s * 0.125f : NEG;
  }
  __syncthreads();
  mx = NEG;
  for (int n = tid; n < NSEQ; n += 256) mx = fmaxf(mx, logits[n]);
  red[tid] = mx; __syncthreads();
  for (int s = 128; s > 0; s >>= 1) { if (tid < s) red[tid] = fmaxf(red[tid], red[tid + s]); __syncthreads(); }
  mx = red[0]; __syncthreads();
  sum = 0.f;
  for (int n = tid; n < NSEQ; n += 256) { float e = __expf(logits[n] - mx); logits[n] = e; sum += e; }
  red[tid] = sum; __syncthreads();
  for (int s = 128; s > 0; s >>= 1) { if (tid < s) red[tid] += red[tid + s]; __syncthreads(); }
  inv = 1.f / red[0]; __syncthreads();
  {
    const int d = tid & 63, grp = tid >> 6;
    float a = 0.f;
    for (int n = grp; n < NSEQ; n += 4) a += logits[n] * kbase[(size_t)n * 64 + d];
    red[tid] = a; __syncthreads();
    if (tid < 64)
      gk[bh * 64 + tid] = gvec[tid] * (red[tid] + red[tid + 64] + red[tid + 128] + red[tid + 192]) * inv;
  }
}

// ---------- kernel: r = (v (.) gk) @ W_r + b_r + q  -> f16, (b,n,h*64+e) layout ----------
__global__ __launch_bounds__(256) void r_gemm(const float* __restrict__ Vf,
                                              const float* __restrict__ Qf,
                                              const float* __restrict__ Wr,
                                              const float* __restrict__ br,
                                              const float* __restrict__ gk,
                                              _Float16* __restrict__ Rh) {
  constexpr int LDA = 68, LDB = 68;          // 136B rows: 8B-aligned for h4 stores
  __shared__ _Float16 As[128 * LDA];
  __shared__ _Float16 Bs[64 * LDB];
  __shared__ float gks[64];
  const int tid  = threadIdx.x;
  const int lane = tid & 31;
  const int wave = tid >> 5;
  const int wm = wave & 3, wn = wave >> 2;
  const int bh   = blockIdx.y;
  const int row0 = blockIdx.x * 128;
  const int b = bh >> 4, h = bh & 15;
  if (tid < 64) gks[tid] = gk[bh * 64 + tid];
  __syncthreads();

  const float* vbase = Vf + (size_t)bh * NSEQ * 64;
  for (int e = tid; e < 128 * 16; e += 256) {     // 16B f32 loads, 8B f16 LDS stores
    const int m = e >> 4, c = e & 15;
    const float4 v = ((const float4*)(vbase + (size_t)(row0 + m) * 64))[c];
    h4 o;
    o[0] = (_Float16)(v.x * gks[c * 4 + 0]);
    o[1] = (_Float16)(v.y * gks[c * 4 + 1]);
    o[2] = (_Float16)(v.z * gks[c * 4 + 2]);
    o[3] = (_Float16)(v.w * gks[c * 4 + 3]);
    *(h4*)&As[m * LDA + c * 4] = o;
  }
  for (int e = tid; e < 64 * 64; e += 256) {
    const int n = e >> 6, d = e & 63;
    Bs[n * LDB + d] = (_Float16)Wr[d * 64 + n];   // W_r is (d_in, e_out)
  }
  __syncthreads();

  v8f acc[2][2] = {};
#pragma unroll
  for (int k0 = 0; k0 < 64; k0 += 32) {
    v16h a0 = load_fragA(&As[(wm * 32) * LDA + k0],      LDA, lane);
    v16h a1 = load_fragA(&As[(wm * 32 + 16) * LDA + k0], LDA, lane);
    v16h b0 = load_fragB(&Bs[(wn * 32) * LDB + k0],      LDB, lane);
    v16h b1 = load_fragB(&Bs[(wn * 32 + 16) * LDB + k0], LDB, lane);
    acc[0][0] = wmma16(a0, b0, acc[0][0]);
    acc[0][1] = wmma16(a0, b1, acc[0][1]);
    acc[1][0] = wmma16(a1, b0, acc[1][0]);
    acc[1][1] = wmma16(a1, b1, acc[1][1]);
  }

  const float* qbase = Qf + (size_t)bh * NSEQ * 64;
#pragma unroll
  for (int sm = 0; sm < 2; ++sm)
#pragma unroll
    for (int sn = 0; sn < 2; ++sn) {
      const int e     = wn * 32 + sn * 16 + (lane & 15);
      const int mbase = row0 + wm * 32 + sm * 16 + ((lane >> 4) << 3);
#pragma unroll
      for (int j = 0; j < 8; ++j) {
        const int mr = mbase + j;
        float val = acc[sm][sn][j] + br[e] + qbase[(size_t)mr * 64 + e];
        Rh[((size_t)(b * NSEQ + mr)) * DIMM + h * 64 + e] = (_Float16)val;
      }
    }
}

// ---------- workspace layout (bytes) ----------
#define OFF_XH     ((size_t)0)                        // 16384*1024 f16 = 33554432
#define OFF_WQKVH  ((size_t)33554432)                 // 1024*3072 f16 =  6291456
#define OFF_WOUTH  ((size_t)39845888)                 // 1024*1024 f16 =  2097152
#define OFF_Q      ((size_t)41943040)                 // 4*16*4096*64 f32 = 67108864
#define OFF_K      ((size_t)109051904)
#define OFF_V      ((size_t)176160768)
#define OFF_GQ     ((size_t)243269632)                // 64*64 f32
#define OFF_GK     ((size_t)243286016)
#define OFF_RH     ((size_t)243302400)                // 16384*1024 f16 = 33554432
// total ~ 276,856,832 bytes

extern "C" void kernel_launch(void* const* d_in, const int* in_sizes, int n_in,
                              void* d_out, int out_size, void* d_ws, size_t ws_size,
                              hipStream_t stream) {
  const float* x    = (const float*)d_in[0];
  const unsigned char* mask = (const unsigned char*)d_in[1];
  const float* Wqkv = (const float*)d_in[2];
  const float* wq   = (const float*)d_in[3];
  const float* wk   = (const float*)d_in[4];
  const float* Wr   = (const float*)d_in[5];
  const float* br   = (const float*)d_in[6];
  const float* Wout = (const float*)d_in[7];
  const float* bout = (const float*)d_in[8];
  float* out = (float*)d_out;

  char* ws = (char*)d_ws;
  _Float16* xh     = (_Float16*)(ws + OFF_XH);
  _Float16* wqkvh  = (_Float16*)(ws + OFF_WQKVH);
  _Float16* wouth  = (_Float16*)(ws + OFF_WOUTH);
  float*    Qf     = (float*)(ws + OFF_Q);
  float*    Kf     = (float*)(ws + OFF_K);
  float*    Vf     = (float*)(ws + OFF_V);
  float*    gq     = (float*)(ws + OFF_GQ);
  float*    gk     = (float*)(ws + OFF_GK);
  _Float16* rh     = (_Float16*)(ws + OFF_RH);

  // 1) precision downcast (one-time; halves streamed bytes for the GEMMs)
  cvt_f32_f16<<<2048, 256, 0, stream>>>((const float4*)x,    (h4*)xh,    (long)NROWS * DIMM / 4);
  cvt_f32_f16<<<512,  256, 0, stream>>>((const float4*)Wqkv, (h4*)wqkvh, (long)DIMM * NC_QKV / 4);
  cvt_f32_f16<<<256,  256, 0, stream>>>((const float4*)Wout, (h4*)wouth, (long)DIMM * DIMM / 4);

  // 2) qkv GEMM (103 GFLOP) + fused head transpose
  qkv_gemm<<<dim3(NROWS / 128, NC_QKV / 128), 256, 0, stream>>>(xh, wqkvh, Qf, Kf, Vf);

  // 3) global-attention reductions (memory-bound, one block per (b,h))
  attn_stats<<<BATCH * HEADS, 256, 0, stream>>>(Qf, Kf, wq, wk, mask, gq, gk);

  // 4) r = (v (.) gk) @ W_r + b_r + q  (WMMA, K=64)
  r_gemm<<<dim3(NSEQ / 128, BATCH * HEADS), 256, 0, stream>>>(Vf, Qf, Wr, br, gk, rh);

  // 5) out = r @ W_out + b_out (34 GFLOP)
  out_gemm<<<dim3(NROWS / 128, DIMM / 128), 256, 0, stream>>>(rh, wouth, bout, out);
}
